// MultiHeadAttention_74878459838739
// MI455X (gfx1250) — matrix operand order repaired and checked by reading, not compiled, not run on hardware
//
#include <hip/hip_runtime.h>

// ---------------------------------------------------------------------------
// MI455X (gfx1250) causal multi-head attention, bf16 WMMA pipeline.
//   B=2, T=2048, C=1024, H=16, D=64
// Stages: cvt(x,W) -> QKV GEMM (V written transposed) -> flash attn -> proj.
// GEMMs: 256x128 workgroup tile, 8 waves (4x2), 64x64 wave tile:
//        16 wmma per 16 ds_load_b128 per K-step (2x better reuse than 64x32).
// ---------------------------------------------------------------------------

typedef __attribute__((ext_vector_type(16))) __bf16       v16bf;
typedef __attribute__((ext_vector_type(8)))  float        v8f;
typedef __attribute__((ext_vector_type(4)))  unsigned int u32x4;

union FragBf {        // 16x32 A-frag or 32x16 B-frag (bf16), 8 VGPRs
    u32x4 q[2];
    v16bf v;
};

__device__ __forceinline__ v8f wmma_bf16(v16bf a, v16bf b, v8f c) {
    return __builtin_amdgcn_wmma_f32_16x16x32_bf16(
        /*neg_a=*/false, a, /*neg_b=*/false, b,
        /*c_mod=*/(short)0, c, /*reuse_a=*/false, /*reuse_b=*/false);
}

__device__ __forceinline__ unsigned short f2bf(float f) {
    unsigned int u = __float_as_uint(f);
    unsigned int r = u + 0x7FFFu + ((u >> 16) & 1u);   // round-nearest-even
    return (unsigned short)(r >> 16);
}

// ---------------------------------------------------------------------------
// Converters
// ---------------------------------------------------------------------------
__global__ void cvt_bf16_kernel(const float* __restrict__ src,
                                unsigned short* __restrict__ dst, int n) {
    int i = blockIdx.x * blockDim.x + threadIdx.x;
    if (i < n) dst[i] = f2bf(src[i]);
}

// W stored (in,out); produce Wt[out][in] bf16 so B-frags read contiguous K.
__global__ void transpose_cvt_kernel(const float* __restrict__ W,
                                     unsigned short* __restrict__ Wt,
                                     int in_dim, int out_dim) {
    int i = blockIdx.x * blockDim.x + threadIdx.x;
    if (i < in_dim * out_dim) {
        int o  = i / in_dim;
        int ii = i - o * in_dim;
        Wt[i] = f2bf(W[(size_t)ii * out_dim + o]);
    }
}

// ---------------------------------------------------------------------------
// QKV GEMM: x[4096x1024] (bf16) @ WqkvT[3072x1024] (bf16) + bias
// Q,K -> [B,H,T,D] bf16 ; V -> [B,H,D,T] bf16 (pre-transposed for attention).
// ---------------------------------------------------------------------------
__global__ __launch_bounds__(256) void gemm_qkv_kernel(
    const unsigned short* __restrict__ xb,
    const unsigned short* __restrict__ wT,
    const float* __restrict__ bias,
    unsigned short* __restrict__ qb,
    unsigned short* __restrict__ kb,
    unsigned short* __restrict__ vt) {
    __shared__ unsigned short sA[256 * 40];   // 256 rows x 32 K (+8 pad)
    __shared__ unsigned short sB[128 * 40];   // 128 cols x 32 K (+8 pad)

    const int tid  = threadIdx.x;
    const int lane = tid & 31, wave = tid >> 5;
    const int wm = wave & 3, wn = wave >> 2;     // 4 x 2 wave grid
    const int ln16 = lane & 15, hf = lane >> 4;
    const int row0 = blockIdx.y * 256;           // token rows
    const int col0 = blockIdx.x * 128;           // out cols (of 3072)

    v8f acc[4][4];
    #pragma unroll
    for (int nt = 0; nt < 4; ++nt) {
        float bv = bias[col0 + wn * 64 + nt * 16 + ln16];
        #pragma unroll
        for (int mt = 0; mt < 4; ++mt)
            #pragma unroll
            for (int r = 0; r < 8; ++r) acc[mt][nt][r] = bv;
    }

    for (int k0 = 0; k0 < 1024; k0 += 32) {
        __syncthreads();
        #pragma unroll
        for (int i = 0; i < 4; ++i) {            // A: 1024 16B chunks
            int idx = tid + i * 256;
            int r = idx >> 2, c8 = (idx & 3) * 8;
            u32x4 va = *reinterpret_cast<const u32x4*>(
                xb + (size_t)(row0 + r) * 1024 + k0 + c8);
            *reinterpret_cast<u32x4*>(sA + r * 40 + c8) = va;
        }
        #pragma unroll
        for (int i = 0; i < 2; ++i) {            // B: 512 16B chunks
            int idx = tid + i * 256;
            int r = idx >> 2, c8 = (idx & 3) * 8;
            u32x4 vb4 = *reinterpret_cast<const u32x4*>(
                wT + (size_t)(col0 + r) * 1024 + k0 + c8);
            *reinterpret_cast<u32x4*>(sB + r * 40 + c8) = vb4;
        }
        __syncthreads();

        FragBf afr[4], bfr[4];
        #pragma unroll
        for (int mt = 0; mt < 4; ++mt) {
            int ar = wm * 64 + mt * 16 + ln16;
            afr[mt].q[0] = *reinterpret_cast<const u32x4*>(sA + ar * 40 + hf * 8);
            afr[mt].q[1] = *reinterpret_cast<const u32x4*>(sA + ar * 40 + 16 + hf * 8);
        }
        #pragma unroll
        for (int nt = 0; nt < 4; ++nt) {
            int br = wn * 64 + nt * 16 + ln16;
            bfr[nt].q[0] = *reinterpret_cast<const u32x4*>(sB + br * 40 + hf * 8);
            bfr[nt].q[1] = *reinterpret_cast<const u32x4*>(sB + br * 40 + 16 + hf * 8);
        }
        #pragma unroll
        for (int mt = 0; mt < 4; ++mt)
            #pragma unroll
            for (int nt = 0; nt < 4; ++nt)
                acc[mt][nt] = wmma_bf16(afr[mt].v, bfr[nt].v, acc[mt][nt]);
    }

    // scatter: q,k -> [B,H,T,D]; v -> [B,H,D,T]
    #pragma unroll
    for (int mt = 0; mt < 4; ++mt)
        #pragma unroll
        for (int nt = 0; nt < 4; ++nt) {
            int colg = col0 + wn * 64 + nt * 16 + ln16;  // 0..3071
            int sec  = colg >> 10;
            int cc   = colg & 1023;
            int head = cc >> 6, d = cc & 63;
            #pragma unroll
            for (int r = 0; r < 8; ++r) {
                int rowg = row0 + wm * 64 + mt * 16 + r + 8 * hf;  // 0..4095
                int b = rowg >> 11, tt = rowg & 2047;
                unsigned short bf = f2bf(acc[mt][nt][r]);
                if (sec == 2) {
                    vt[((size_t)((b * 16 + head) * 64 + d)) * 2048 + tt] = bf;
                } else {
                    unsigned short* dst = (sec == 0) ? qb : kb;
                    dst[((size_t)((b * 16 + head) * 2048 + tt)) * 64 + d] = bf;
                }
            }
        }
}

// ---------------------------------------------------------------------------
// Flash attention: grid (T/128, B*H); 8 waves x 16 query rows each.
// 32-key tiles; S=Q*K^T (4 wmma), online softmax, O+=P*V (4 wmma).
// V arrives pre-transposed [B,H,D,T] -> LDS staging is pure b128 copies.
// ---------------------------------------------------------------------------
__global__ __launch_bounds__(256) void attn_kernel(
    const unsigned short* __restrict__ qg,
    const unsigned short* __restrict__ kg,
    const unsigned short* __restrict__ vtg,
    unsigned short* __restrict__ yb) {
    __shared__ unsigned short sK [32 * 72];      // [key][d] 32x64 (+8 pad)
    __shared__ unsigned short sVt[64 * 40];      // [d][key] 64x32 (+8 pad)
    __shared__ unsigned short sP [8 * 16 * 40];  // per-wave P patch 16x32

    const int tid  = threadIdx.x;
    const int lane = tid & 31, wave = tid >> 5;
    const int ln16 = lane & 15, hf = lane >> 4;
    const int bh = blockIdx.y;                   // 0..31
    const int b = bh >> 4, h = bh & 15;
    const int qblk = blockIdx.x;                 // 0..15
    const size_t base = (size_t)bh * 2048 * 64;
    const unsigned short* Q  = qg  + base;
    const unsigned short* K  = kg  + base;
    const unsigned short* Vt = vtg + base;       // [d][t]
    const int qrow0 = qblk * 128 + wave * 16;

    // Q fragments: 16 rows x 64 K  ->  two 16x32 A-frags
    FragBf qf[2];
    {
        const unsigned short* qr = Q + (size_t)(qrow0 + ln16) * 64;
        qf[0].q[0] = *reinterpret_cast<const u32x4*>(qr + hf * 8);
        qf[0].q[1] = *reinterpret_cast<const u32x4*>(qr + 16 + hf * 8);
        qf[1].q[0] = *reinterpret_cast<const u32x4*>(qr + 32 + hf * 8);
        qf[1].q[1] = *reinterpret_cast<const u32x4*>(qr + 48 + hf * 8);
    }

    v8f o[4];
    #pragma unroll
    for (int dt = 0; dt < 4; ++dt)
        #pragma unroll
        for (int r = 0; r < 8; ++r) o[dt][r] = 0.f;
    float m[8], l[8];
    #pragma unroll
    for (int r = 0; r < 8; ++r) { m[r] = -1e30f; l[r] = 0.f; }

    const int kmax = qblk * 128 + 128;           // workgroup-wide bound
    for (int kt = 0; kt < kmax; kt += 32) {
        __syncthreads();
        {   // stage K tile [32 keys][64 d] : 256 chunks of 16B
            int key = tid >> 3, s8 = (tid & 7) * 8;
            u32x4 kv = *reinterpret_cast<const u32x4*>(K + (size_t)(kt + key) * 64 + s8);
            *reinterpret_cast<u32x4*>(sK + key * 72 + s8) = kv;
        }
        {   // stage Vt tile [64 d][32 keys] : 256 chunks of 16B
            int dRow = tid >> 2, k8 = (tid & 3) * 8;
            u32x4 vv = *reinterpret_cast<const u32x4*>(Vt + (size_t)dRow * 2048 + kt + k8);
            *reinterpret_cast<u32x4*>(sVt + dRow * 40 + k8) = vv;
        }
        __syncthreads();

        if (kt <= qrow0 + 15) {                  // causal: wave has work here
            v8f s[2];
            #pragma unroll
            for (int st = 0; st < 2; ++st) {
                FragBf bf0, bf1;                 // K^T 64x16 B-frags (k-split)
                int kr = (st * 16 + ln16) * 72;
                bf0.q[0] = *reinterpret_cast<const u32x4*>(sK + kr + hf * 8);
                bf0.q[1] = *reinterpret_cast<const u32x4*>(sK + kr + 16 + hf * 8);
                bf1.q[0] = *reinterpret_cast<const u32x4*>(sK + kr + 32 + hf * 8);
                bf1.q[1] = *reinterpret_cast<const u32x4*>(sK + kr + 48 + hf * 8);
                v8f c = {0.f, 0.f, 0.f, 0.f, 0.f, 0.f, 0.f, 0.f};
                c = wmma_bf16(qf[0].v, bf0.v, c);
                c = wmma_bf16(qf[1].v, bf1.v, c);
                #pragma unroll
                for (int r = 0; r < 8; ++r) {
                    float sv = c[r] * 0.125f;    // 1/sqrt(64)
                    int qi = qrow0 + r + 8 * hf;
                    int ki = kt + st * 16 + ln16;
                    s[st][r] = (ki > qi) ? -1e30f : sv;
                }
            }
            // online softmax (row stats across 16-lane groups)
            float sc[8];
            #pragma unroll
            for (int r = 0; r < 8; ++r) {
                float mx = fmaxf(s[0][r], s[1][r]);
                #pragma unroll
                for (int off = 8; off >= 1; off >>= 1)
                    mx = fmaxf(mx, __shfl_xor(mx, off, 16));
                float mn = fmaxf(m[r], mx);
                sc[r] = __expf(m[r] - mn);
                float p0 = __expf(s[0][r] - mn);
                float p1 = __expf(s[1][r] - mn);
                s[0][r] = p0; s[1][r] = p1;
                float sum = p0 + p1;
                #pragma unroll
                for (int off = 8; off >= 1; off >>= 1)
                    sum += __shfl_xor(sum, off, 16);
                l[r] = l[r] * sc[r] + sum;
                m[r] = mn;
            }
            #pragma unroll
            for (int dt = 0; dt < 4; ++dt)
                #pragma unroll
                for (int r = 0; r < 8; ++r) o[dt][r] *= sc[r];

            // P (C-layout f32) -> per-wave LDS patch -> A-frag bf16
            unsigned short* wp = sP + wave * 16 * 40;
            #pragma unroll
            for (int st = 0; st < 2; ++st)
                #pragma unroll
                for (int r = 0; r < 8; ++r)
                    wp[(r + 8 * hf) * 40 + st * 16 + ln16] = f2bf(s[st][r]);
            FragBf pf;
            pf.q[0] = *reinterpret_cast<const u32x4*>(wp + ln16 * 40 + hf * 8);
            pf.q[1] = *reinterpret_cast<const u32x4*>(wp + ln16 * 40 + 16 + hf * 8);

            #pragma unroll
            for (int dt = 0; dt < 4; ++dt) {     // O += P * V
                FragBf vf;                       // V 32x16 B-frag from sVt
                int dr = (dt * 16 + ln16) * 40;
                vf.q[0] = *reinterpret_cast<const u32x4*>(sVt + dr + hf * 8);
                vf.q[1] = *reinterpret_cast<const u32x4*>(sVt + dr + 16 + hf * 8);
                o[dt] = wmma_bf16(pf.v, vf.v, o[dt]);
            }
        }
    }

    // normalize + write y[b][t][h*64+d] bf16
    #pragma unroll
    for (int r = 0; r < 8; ++r) {
        float inv = 1.0f / l[r];
        int tok = b * 2048 + qrow0 + r + 8 * hf;
        #pragma unroll
        for (int dt = 0; dt < 4; ++dt) {
            int d = dt * 16 + ln16;
            yb[(size_t)tok * 1024 + h * 64 + d] = f2bf(o[dt][r] * inv);
        }
    }
}

// ---------------------------------------------------------------------------
// Output projection: y[4096x1024] @ WprojT[1024x1024] + bias -> f32 out
// ---------------------------------------------------------------------------
__global__ __launch_bounds__(256) void gemm_proj_kernel(
    const unsigned short* __restrict__ yb,
    const unsigned short* __restrict__ wT,
    const float* __restrict__ bias,
    float* __restrict__ out) {
    __shared__ unsigned short sA[256 * 40];
    __shared__ unsigned short sB[128 * 40];

    const int tid  = threadIdx.x;
    const int lane = tid & 31, wave = tid >> 5;
    const int wm = wave & 3, wn = wave >> 2;
    const int ln16 = lane & 15, hf = lane >> 4;
    const int row0 = blockIdx.y * 256;
    const int col0 = blockIdx.x * 128;

    v8f acc[4][4];
    #pragma unroll
    for (int nt = 0; nt < 4; ++nt) {
        float bv = bias[col0 + wn * 64 + nt * 16 + ln16];
        #pragma unroll
        for (int mt = 0; mt < 4; ++mt)
            #pragma unroll
            for (int r = 0; r < 8; ++r) acc[mt][nt][r] = bv;
    }

    for (int k0 = 0; k0 < 1024; k0 += 32) {
        __syncthreads();
        #pragma unroll
        for (int i = 0; i < 4; ++i) {
            int idx = tid + i * 256;
            int r = idx >> 2, c8 = (idx & 3) * 8;
            u32x4 va = *reinterpret_cast<const u32x4*>(
                yb + (size_t)(row0 + r) * 1024 + k0 + c8);
            *reinterpret_cast<u32x4*>(sA + r * 40 + c8) = va;
        }
        #pragma unroll
        for (int i = 0; i < 2; ++i) {
            int idx = tid + i * 256;
            int r = idx >> 2, c8 = (idx & 3) * 8;
            u32x4 vb4 = *reinterpret_cast<const u32x4*>(
                wT + (size_t)(col0 + r) * 1024 + k0 + c8);
            *reinterpret_cast<u32x4*>(sB + r * 40 + c8) = vb4;
        }
        __syncthreads();

        FragBf afr[4], bfr[4];
        #pragma unroll
        for (int mt = 0; mt < 4; ++mt) {
            int ar = wm * 64 + mt * 16 + ln16;
            afr[mt].q[0] = *reinterpret_cast<const u32x4*>(sA + ar * 40 + hf * 8);
            afr[mt].q[1] = *reinterpret_cast<const u32x4*>(sA + ar * 40 + 16 + hf * 8);
        }
        #pragma unroll
        for (int nt = 0; nt < 4; ++nt) {
            int br = wn * 64 + nt * 16 + ln16;
            bfr[nt].q[0] = *reinterpret_cast<const u32x4*>(sB + br * 40 + hf * 8);
            bfr[nt].q[1] = *reinterpret_cast<const u32x4*>(sB + br * 40 + 16 + hf * 8);
        }
        #pragma unroll
        for (int mt = 0; mt < 4; ++mt)
            #pragma unroll
            for (int nt = 0; nt < 4; ++nt)
                acc[mt][nt] = wmma_bf16(afr[mt].v, bfr[nt].v, acc[mt][nt]);
    }

    #pragma unroll
    for (int mt = 0; mt < 4; ++mt)
        #pragma unroll
        for (int nt = 0; nt < 4; ++nt) {
            int colg = col0 + wn * 64 + nt * 16 + ln16;
            #pragma unroll
            for (int r = 0; r < 8; ++r) {
                int rowg = row0 + wm * 64 + mt * 16 + r + 8 * hf;
                out[(size_t)rowg * 1024 + colg] = acc[mt][nt][r];
            }
        }
}

// ---------------------------------------------------------------------------
extern "C" void kernel_launch(void* const* d_in, const int* in_sizes, int n_in,
                              void* d_out, int out_size, void* d_ws, size_t ws_size,
                              hipStream_t stream) {
    (void)in_sizes; (void)n_in; (void)out_size; (void)ws_size;
    const float* x      = (const float*)d_in[0];
    const float* W_qkv  = (const float*)d_in[1];
    const float* b_qkv  = (const float*)d_in[2];
    const float* W_proj = (const float*)d_in[3];
    const float* b_proj = (const float*)d_in[4];
    float* out = (float*)d_out;

    char* ws = (char*)d_ws;
    unsigned short* xb     = (unsigned short*)ws;  ws += (size_t)4096 * 1024 * 2;
    unsigned short* wqkvT  = (unsigned short*)ws;  ws += (size_t)3072 * 1024 * 2;
    unsigned short* wprojT = (unsigned short*)ws;  ws += (size_t)1024 * 1024 * 2;
    unsigned short* qb     = (unsigned short*)ws;  ws += (size_t)32 * 2048 * 64 * 2;
    unsigned short* kb     = (unsigned short*)ws;  ws += (size_t)32 * 2048 * 64 * 2;
    unsigned short* vt     = (unsigned short*)ws;  ws += (size_t)32 * 2048 * 64 * 2;
    unsigned short* yb     = (unsigned short*)ws;  ws += (size_t)4096 * 1024 * 2;

    cvt_bf16_kernel<<<(4096 * 1024 + 255) / 256, 256, 0, stream>>>(x, xb, 4096 * 1024);
    transpose_cvt_kernel<<<(3072 * 1024 + 255) / 256, 256, 0, stream>>>(W_qkv, wqkvT, 1024, 3072);
    transpose_cvt_kernel<<<(1024 * 1024 + 255) / 256, 256, 0, stream>>>(W_proj, wprojT, 1024, 1024);

    gemm_qkv_kernel<<<dim3(24, 16), 256, 0, stream>>>(xb, wqkvT, b_qkv, qb, kb, vt);
    attn_kernel    <<<dim3(16, 32), 256, 0, stream>>>(qb, kb, vt, yb);
    gemm_proj_kernel<<<dim3(8, 16), 256, 0, stream>>>(yb, wprojT, b_proj, out);
}
